// AttentionalPooler_2671469658384
// MI455X (gfx1250) — compile-verified, matrix operand order checked by reading, not run on hardware
//
#include <hip/hip_runtime.h>

#define D_MODEL   768
#define CTX_DIM   1024
#define N_HEAD    12
#define N_QUERIES 256
#define DIM_HEAD  64
#define BATCH     8
#define SEQ       4096
#define KV_W      (2*D_MODEL)

typedef __attribute__((ext_vector_type(16))) __bf16 v16bf;
typedef __attribute__((ext_vector_type(8)))  float  v8f;
typedef unsigned int v4u __attribute__((ext_vector_type(4)));
typedef unsigned int v8u __attribute__((ext_vector_type(8)));

union Frag { v16bf v; uint4 u[2]; };

__device__ __forceinline__ unsigned short f2bf(float f) {
  unsigned int u = __float_as_uint(f);
  u += 0x7FFFu + ((u >> 16) & 1u);            // round-to-nearest-even
  return (unsigned short)(u >> 16);
}
__device__ __forceinline__ v8f wmma_bf16(Frag a, Frag b, v8f c) {
  return __builtin_amdgcn_wmma_f32_16x16x32_bf16(false, a.v, false, b.v,
                                                 (short)0, c, false, false);
}

// low 32 bits of a generic pointer to LDS == LDS byte address (aperture rule)
__device__ __forceinline__ unsigned lds_addr(const void* p) {
  return (unsigned)(size_t)p;
}
// CDNA5 async global->LDS copy, tracked by ASYNCcnt (no VGPR data path)
__device__ __forceinline__ void async_b128(unsigned lds, const void* g) {
  asm volatile("global_load_async_to_lds_b128 %0, %1, off"
               :: "v"(lds), "v"((unsigned long long)(size_t)g)
               : "memory");
}
__device__ __forceinline__ void wait_async0() {
  asm volatile("s_wait_asynccnt 0x0" ::: "memory");
}
__device__ __forceinline__ void wait_async2() {   // keep newest 2 in flight
  asm volatile("s_wait_asynccnt 0x2" ::: "memory");
}
__device__ __forceinline__ void wait_async4() {   // keep newest 4 in flight
  asm volatile("s_wait_asynccnt 0x4" ::: "memory");
}

// ---------------------------------------------------------------------------
// TDM: DMA a 2-D tile (tile_rows x tile_cols bf16, row stride in elements)
// from global memory into contiguous LDS. D# groups built in SGPR tuples.
// Issue from ONE wave only; tracked by TENSORcnt.
// ---------------------------------------------------------------------------
__device__ __forceinline__ void tdm_load_2d(const void* g, unsigned lds,
                                            unsigned tile_rows,
                                            unsigned tile_cols,
                                            unsigned row_stride) {
  unsigned long long ga = (unsigned long long)(size_t)g;
  v4u g0;
  g0[0] = 1u;                                           // count=1, user mode
  g0[1] = lds;                                          // lds_addr (bytes)
  g0[2] = (unsigned)ga;                                 // global_addr[31:0]
  g0[3] = ((unsigned)(ga >> 32) & 0x01FFFFFFu)          // global_addr[56:32]
          | 0x80000000u;                                // type = 2 ("image")
  const unsigned td0 = 0x40000000u, td1 = 0x40000000u;  // huge dims: no OOB
  v8u g1;
  g1[0] = 0x00010000u;                      // wg_mask=0, data_size=1 (2 bytes)
  g1[1] = (td0 & 0xFFFFu) << 16;            // tensor_dim0[15:0]
  g1[2] = (td0 >> 16) | ((td1 & 0xFFFFu) << 16);
  g1[3] = (td1 >> 16) | (tile_cols << 16);  // tile_dim0
  g1[4] = tile_rows & 0xFFFFu;              // tile_dim1 (tile_dim2 = 0)
  g1[5] = row_stride;                       // tensor_dim0_stride[31:0]
  g1[6] = 0u;
  g1[7] = 0u;
  asm volatile("tensor_load_to_lds %0, %1" :: "s"(g0), "s"(g1) : "memory");
}

// ---------------------------------------------------------------------------
// Kernel 1: per-row mean / rstd of x (rows of length CTX_DIM)
// ---------------------------------------------------------------------------
__global__ void rowstats_kernel(const float* __restrict__ x,
                                float* __restrict__ stats) {
  const int row = blockIdx.x;
  const float* xr = x + (size_t)row * CTX_DIM;
  float s = 0.f, ss = 0.f;
  for (int i = threadIdx.x; i < CTX_DIM; i += 256) {
    float v = xr[i]; s += v; ss += v * v;
  }
  for (int off = 16; off > 0; off >>= 1) {
    s  += __shfl_xor(s,  off, 32);
    ss += __shfl_xor(ss, off, 32);
  }
  __shared__ float red[16];
  const int wid = threadIdx.x >> 5;
  if ((threadIdx.x & 31) == 0) { red[wid] = s; red[8 + wid] = ss; }
  __syncthreads();
  if (threadIdx.x == 0) {
    float S = 0.f, SS = 0.f;
    for (int i = 0; i < 8; ++i) { S += red[i]; SS += red[8 + i]; }
    float mu  = S / (float)CTX_DIM;
    float var = SS / (float)CTX_DIM - mu * mu;
    stats[2 * row]     = mu;
    stats[2 * row + 1] = rsqrtf(var + 1e-5f);
  }
}

// ---------------------------------------------------------------------------
// Kernel 2: layernorm(query) -> bf16   (one row of D_MODEL per block)
// ---------------------------------------------------------------------------
__global__ void qln_kernel(const float* __restrict__ q,
                           const float* __restrict__ w,
                           const float* __restrict__ b,
                           unsigned short* __restrict__ out) {
  const int row = blockIdx.x;
  const float* qr = q + (size_t)row * D_MODEL;
  float s = 0.f, ss = 0.f;
  for (int i = threadIdx.x; i < D_MODEL; i += 256) {
    float v = qr[i]; s += v; ss += v * v;
  }
  for (int off = 16; off > 0; off >>= 1) {
    s  += __shfl_xor(s,  off, 32);
    ss += __shfl_xor(ss, off, 32);
  }
  __shared__ float red[16];
  __shared__ float mu_s, rs_s;
  const int wid = threadIdx.x >> 5;
  if ((threadIdx.x & 31) == 0) { red[wid] = s; red[8 + wid] = ss; }
  __syncthreads();
  if (threadIdx.x == 0) {
    float S = 0.f, SS = 0.f;
    for (int i = 0; i < 8; ++i) { S += red[i]; SS += red[8 + i]; }
    float mu  = S / (float)D_MODEL;
    float var = SS / (float)D_MODEL - mu * mu;
    mu_s = mu; rs_s = rsqrtf(var + 1e-5f);
  }
  __syncthreads();
  const float mu = mu_s, rs = rs_s;
  for (int i = threadIdx.x; i < D_MODEL; i += 256)
    out[(size_t)row * D_MODEL + i] = f2bf((qr[i] - mu) * rs * w[i] + b[i]);
}

// ---------------------------------------------------------------------------
// Kernel 3: fp32 -> bf16 convert WITH transpose: in[K][N] -> out[N][K]
// ---------------------------------------------------------------------------
__global__ void cvt_t_kernel(const float* __restrict__ in,
                             unsigned short* __restrict__ out, int K, int N) {
  int i = blockIdx.x * 256 + threadIdx.x;
  if (i < K * N) {
    int k = i / N, n = i % N;
    out[(size_t)n * K + k] = f2bf(in[i]);
  }
}

// ---------------------------------------------------------------------------
// Generic bf16 GEMM: C[M,N] = A[M,K] * Bt[N,K]^T.  128x128 block, BK=32.
// Double-buffered LDS tiles staged by the Tensor Data Mover (wave 0 issues,
// TENSORcnt<=2 keeps the next tile's two DMAs in flight during compute).
// ---------------------------------------------------------------------------
template <bool OUT_F32>
__global__ void gemm_bf16_kernel(const unsigned short* __restrict__ A,
                                 const unsigned short* __restrict__ Bt,
                                 void* __restrict__ Cout,
                                 int M, int N, int K) {
  __shared__ unsigned short As[2][128 * 32];   // As[buf][m][k]
  __shared__ unsigned short Bs[2][128 * 32];   // Bs[buf][n][k]
  const int tid = threadIdx.x, lane = tid & 31, w = tid >> 5;
  const int bm = blockIdx.x * 128, bn = blockIdx.y * 128;
  const int wr = w & 3, wc = w >> 2;
  const int fm = lane & 15, fk = (lane >> 4) * 8;

  v8f zero = {};
  v8f acc[2][4];
#pragma unroll
  for (int p = 0; p < 2; ++p)
#pragma unroll
    for (int t = 0; t < 4; ++t) acc[p][t] = zero;

  const unsigned short* abase = A  + (size_t)bm * K;
  const unsigned short* bbase = Bt + (size_t)bn * K;

  if (w == 0) {   // prologue: DMA first tiles into buffer 0
    tdm_load_2d(abase, lds_addr(&As[0][0]), 128, 32, K);
    tdm_load_2d(bbase, lds_addr(&Bs[0][0]), 128, 32, K);
  }

  const int nIter = K / 32;
  for (int i = 0; i < nIter; ++i) {
    const int cur = i & 1;
    const bool nxt = (i + 1) < nIter;
    if (w == 0) {
      if (nxt) {
        const int k1 = (i + 1) * 32;
        tdm_load_2d(abase + k1, lds_addr(&As[cur ^ 1][0]), 128, 32, K);
        tdm_load_2d(bbase + k1, lds_addr(&Bs[cur ^ 1][0]), 128, 32, K);
        __builtin_amdgcn_s_wait_tensorcnt(2);   // current buffer resident
      } else {
        __builtin_amdgcn_s_wait_tensorcnt(0);
      }
    }
    __syncthreads();

    Frag a[2], bf[4];
#pragma unroll
    for (int p = 0; p < 2; ++p) {
      const unsigned short* ar = &As[cur][(wr * 32 + p * 16 + fm) * 32];
      a[p].u[0] = *(const uint4*)&ar[fk];
      a[p].u[1] = *(const uint4*)&ar[fk + 16];
    }
#pragma unroll
    for (int t = 0; t < 4; ++t) {
      const unsigned short* br = &Bs[cur][(wc * 64 + t * 16 + fm) * 32];
      bf[t].u[0] = *(const uint4*)&br[fk];
      bf[t].u[1] = *(const uint4*)&br[fk + 16];
    }
#pragma unroll
    for (int p = 0; p < 2; ++p)
#pragma unroll
      for (int t = 0; t < 4; ++t) acc[p][t] = wmma_bf16(a[p], bf[t], acc[p][t]);
    __syncthreads();
  }

  const int cn = lane & 15, rb = (lane >> 4) * 8;
#pragma unroll
  for (int p = 0; p < 2; ++p)
#pragma unroll
    for (int t = 0; t < 4; ++t)
#pragma unroll
      for (int r = 0; r < 8; ++r) {
        size_t gm  = (size_t)(bm + wr * 32 + p * 16 + rb + r);
        size_t idx = gm * (size_t)N + bn + wc * 64 + t * 16 + cn;
        if (OUT_F32) ((float*)Cout)[idx] = acc[p][t][r];
        else         ((unsigned short*)Cout)[idx] = f2bf(acc[p][t][r]);
      }
}

// ---------------------------------------------------------------------------
// kv GEMM, LN fused into A-tile: [kbuf | vtbuf] = LN(X) @ WkvT^T.
// Double-buffered: B tile via async copies, A tile (LN+cvt) pipelined one
// K-chunk ahead through LDS. K half -> kbuf[row][768]; V half -> vtbuf
// [b][h][d][4096] (pre-transposed for the attention kernel).
// ---------------------------------------------------------------------------
__global__ void gemm_lnx_kernel(const float* __restrict__ X,
                                const float* __restrict__ stats,
                                const float* __restrict__ lw,
                                const float* __restrict__ lb,
                                const unsigned short* __restrict__ Bt,
                                unsigned short* __restrict__ kbuf,
                                unsigned short* __restrict__ vtbuf) {
  const int K = CTX_DIM;
  __shared__ unsigned short As[2][128 * 32];
  __shared__ unsigned short Bs[2][128 * 32];
  const int tid = threadIdx.x, lane = tid & 31, w = tid >> 5;
  const int bm = blockIdx.x * 128, bn = blockIdx.y * 128;
  const int wr = w & 3, wc = w >> 2;
  const int lrow = tid >> 1, lkg = (tid & 1) * 16;
  const int fm = lane & 15, fk = (lane >> 4) * 8;

  v8f zero = {};
  v8f acc[2][4];
#pragma unroll
  for (int p = 0; p < 2; ++p)
#pragma unroll
    for (int t = 0; t < 4; ++t) acc[p][t] = zero;

  const float mu = stats[2 * (bm + lrow)];
  const float rs = stats[2 * (bm + lrow) + 1];
  const float* xp = X + (size_t)(bm + lrow) * K + lkg;
  const unsigned short* bp = Bt + (size_t)(bn + lrow) * K + lkg;
  const unsigned bs0 = lds_addr(&Bs[0][lrow * 32 + lkg]);
  const unsigned bs1 = lds_addr(&Bs[1][lrow * 32 + lkg]);

  // prologue: stage chunk 0 into buffer 0
  async_b128(bs0,      bp);
  async_b128(bs0 + 16, bp + 8);
#pragma unroll
  for (int i = 0; i < 16; ++i) {
    float xv = xp[i];
    As[0][lrow * 32 + lkg + i] =
        f2bf((xv - mu) * rs * lw[lkg + i] + lb[lkg + i]);
  }

  const int nIter = K / 32;
  for (int it = 0; it < nIter; ++it) {
    const int cur = it & 1;
    const bool nxt = (it + 1) < nIter;
    if (nxt) {                         // pipeline next chunk into other buffer
      const int k1 = (it + 1) * 32;
      const unsigned bsd = cur ? bs0 : bs1;
      async_b128(bsd,      bp + k1);
      async_b128(bsd + 16, bp + k1 + 8);
      if (it + 2 < nIter) __builtin_prefetch(xp + k1 + 32, 0, 0);
#pragma unroll
      for (int i = 0; i < 16; ++i) {
        float xv = xp[k1 + i];
        As[cur ^ 1][lrow * 32 + lkg + i] =
            f2bf((xv - mu) * rs * lw[k1 + lkg + i] + lb[k1 + lkg + i]);
      }
      wait_async2();
    } else {
      wait_async0();
    }
    __syncthreads();

    Frag a[2], bf[4];
#pragma unroll
    for (int p = 0; p < 2; ++p) {
      const unsigned short* ar = &As[cur][(wr * 32 + p * 16 + fm) * 32];
      a[p].u[0] = *(const uint4*)&ar[fk];
      a[p].u[1] = *(const uint4*)&ar[fk + 16];
    }
#pragma unroll
    for (int t = 0; t < 4; ++t) {
      const unsigned short* br = &Bs[cur][(wc * 64 + t * 16 + fm) * 32];
      bf[t].u[0] = *(const uint4*)&br[fk];
      bf[t].u[1] = *(const uint4*)&br[fk + 16];
    }
#pragma unroll
    for (int p = 0; p < 2; ++p)
#pragma unroll
      for (int t = 0; t < 4; ++t) acc[p][t] = wmma_bf16(a[p], bf[t], acc[p][t]);
    __syncthreads();
  }

  const int cn = lane & 15, rb = (lane >> 4) * 8;
  if (bn < D_MODEL) {           // K half (block-uniform branch)
#pragma unroll
    for (int p = 0; p < 2; ++p)
#pragma unroll
      for (int t = 0; t < 4; ++t)
#pragma unroll
        for (int r = 0; r < 8; ++r) {
          size_t gm = (size_t)(bm + wr * 32 + p * 16 + rb + r);
          int col = bn + wc * 64 + t * 16 + cn;
          kbuf[gm * D_MODEL + col] = f2bf(acc[p][t][r]);
        }
  } else {                      // V half -> transposed layout
#pragma unroll
    for (int p = 0; p < 2; ++p)
#pragma unroll
      for (int t = 0; t < 4; ++t)
#pragma unroll
        for (int r = 0; r < 8; ++r) {
          int gm   = bm + wr * 32 + p * 16 + rb + r;
          int vcol = bn - D_MODEL + wc * 64 + t * 16 + cn;
          int hh = vcol >> 6, d = vcol & 63;
          int bb = gm >> 12, j = gm & (SEQ - 1);
          vtbuf[(((size_t)bb * N_HEAD + hh) * DIM_HEAD + d) * SEQ + j] =
              f2bf(acc[p][t][r]);
        }
  }
}

// ---------------------------------------------------------------------------
// Flash attention: grid (BATCH*N_HEAD, 2). 256 thr = 8 waves; wave w owns
// 16 query rows. Double-buffered K/V tiles via async copies (V pre-transposed)
// with s_wait_asynccnt 0x4 overlapping next-tile loads with WMMA compute.
// ---------------------------------------------------------------------------
__global__ void attn_kernel(const unsigned short* __restrict__ qproj,
                            const unsigned short* __restrict__ kbuf,
                            const unsigned short* __restrict__ vtbuf,
                            unsigned short* __restrict__ attn_out) {
  const int bh = blockIdx.x;
  const int b  = bh / N_HEAD, h = bh % N_HEAD;
  const int q0 = blockIdx.y * 128;

  __shared__ unsigned short qs[128 * 64];        // qs[q][d]
  __shared__ unsigned short ks[2][64 * 64];      // ks[buf][j][d]
  __shared__ unsigned short vts[2][64 * 64];     // vts[buf][d][j]
  __shared__ unsigned short ps[8][16 * 64];      // per-wave P scratch

  const int tid = threadIdx.x, lane = tid & 31, w = tid >> 5;
  const int fm = lane & 15, fk = (lane >> 4) * 8;

  // stage Q tile (async straight copy)
  {
    int r  = tid >> 1;
    int dg = (tid & 1) * 32;
    const unsigned short* src = qproj + (size_t)(q0 + r) * D_MODEL +
                                h * DIM_HEAD + dg;
    unsigned dst = lds_addr(&qs[r * 64 + dg]);
    async_b128(dst,      src);
    async_b128(dst + 16, src + 8);
    async_b128(dst + 32, src + 16);
    async_b128(dst + 48, src + 24);
  }

  const int rr = tid >> 2;              // staging row 0..63
  const int cg = (tid & 3) * 16;        // 16-element column group
  const unsigned short* kcol = kbuf + (size_t)(b * SEQ + rr) * D_MODEL +
                               h * DIM_HEAD + cg;
  const unsigned short* vrow = vtbuf +
      (((size_t)b * N_HEAD + h) * DIM_HEAD + rr) * SEQ + cg;
  const unsigned kd[2] = { lds_addr(&ks[0][rr * 64 + cg]),
                           lds_addr(&ks[1][rr * 64 + cg]) };
  const unsigned vd[2] = { lds_addr(&vts[0][rr * 64 + cg]),
                           lds_addr(&vts[1][rr * 64 + cg]) };

  // prologue: stage first K/V tile into buffer 0
  async_b128(kd[0],      kcol);
  async_b128(kd[0] + 16, kcol + 8);
  async_b128(vd[0],      vrow);
  async_b128(vd[0] + 16, vrow + 8);

  wait_async4();            // Q resident (4 newest = first K/V tile in flight)
  __syncthreads();

  Frag qa0, qa1;
  {
    const unsigned short* qr = &qs[(w * 16 + fm) * 64];
    qa0.u[0] = *(const uint4*)&qr[fk];
    qa0.u[1] = *(const uint4*)&qr[fk + 16];
    qa1.u[0] = *(const uint4*)&qr[32 + fk];
    qa1.u[1] = *(const uint4*)&qr[32 + fk + 16];
  }

  float m_run[8], l_run[8];
#pragma unroll
  for (int r = 0; r < 8; ++r) { m_run[r] = -3.0e38f; l_run[r] = 0.f; }
  v8f zero = {};
  v8f o[4] = {zero, zero, zero, zero};

  for (int j0 = 0; j0 < SEQ; j0 += 64) {
    const int cur = (j0 >> 6) & 1;
    const bool nxt = (j0 + 64) < SEQ;
    if (nxt) {                         // stage next tile into other buffer
      const unsigned short* kp = kcol + (size_t)(j0 + 64) * D_MODEL;
      const unsigned short* vp = vrow + j0 + 64;
      async_b128(kd[cur ^ 1],      kp);
      async_b128(kd[cur ^ 1] + 16, kp + 8);
      async_b128(vd[cur ^ 1],      vp);
      async_b128(vd[cur ^ 1] + 16, vp + 8);
      wait_async4();                   // current tile resident
    } else {
      wait_async0();
    }
    __syncthreads();

    // S = Q * K^T  (4 tiles of 16 keys, 2 WMMAs each over d=64)
    v8f s[4];
#pragma unroll
    for (int t = 0; t < 4; ++t) {
      Frag b0, b1;
      const unsigned short* kr = &ks[cur][(t * 16 + fm) * 64];
      b0.u[0] = *(const uint4*)&kr[fk];
      b0.u[1] = *(const uint4*)&kr[fk + 16];
      b1.u[0] = *(const uint4*)&kr[32 + fk];
      b1.u[1] = *(const uint4*)&kr[32 + fk + 16];
      v8f z = {};
      z = wmma_bf16(qa0, b0, z);
      z = wmma_bf16(qa1, b1, z);
      s[t] = z;
    }

    // online softmax (rows split across 16-lane halves per C/D layout)
#pragma unroll
    for (int r = 0; r < 8; ++r) {
      float mx = fmaxf(fmaxf(s[0][r], s[1][r]), fmaxf(s[2][r], s[3][r]));
#pragma unroll
      for (int off = 1; off < 16; off <<= 1)
        mx = fmaxf(mx, __shfl_xor(mx, off, 32));
      mx *= 0.125f;                               // scale = dh^-0.5
      float nm   = fmaxf(m_run[r], mx);
      float corr = __expf(m_run[r] - nm);
      m_run[r]   = nm;
      float rsum = 0.f;
#pragma unroll
      for (int t = 0; t < 4; ++t) {
        float p = __expf(s[t][r] * 0.125f - nm);
        s[t][r] = p;
        rsum += p;
      }
#pragma unroll
      for (int off = 1; off < 16; off <<= 1)
        rsum += __shfl_xor(rsum, off, 32);
      l_run[r] = l_run[r] * corr + rsum;
#pragma unroll
      for (int t = 0; t < 4; ++t) o[t][r] *= corr;
    }

    // D-layout P -> A-layout via per-wave LDS slab
    {
      unsigned short* pw = &ps[w][0];
      const int rb = (lane >> 4) * 8;
#pragma unroll
      for (int t = 0; t < 4; ++t)
#pragma unroll
        for (int r = 0; r < 8; ++r)
          pw[(rb + r) * 64 + t * 16 + (lane & 15)] = f2bf(s[t][r]);

      Frag p0, p1;
      const unsigned short* pr = &pw[fm * 64];
      p0.u[0] = *(const uint4*)&pr[fk];
      p0.u[1] = *(const uint4*)&pr[fk + 16];
      p1.u[0] = *(const uint4*)&pr[32 + fk];
      p1.u[1] = *(const uint4*)&pr[32 + fk + 16];

      // O += P * V   (4 d-tiles of 16, 2 WMMAs each over j=64)
#pragma unroll
      for (int t = 0; t < 4; ++t) {
        Frag v0, v1;
        const unsigned short* vr = &vts[cur][(t * 16 + fm) * 64];
        v0.u[0] = *(const uint4*)&vr[fk];
        v0.u[1] = *(const uint4*)&vr[fk + 16];
        v1.u[0] = *(const uint4*)&vr[32 + fk];
        v1.u[1] = *(const uint4*)&vr[32 + fk + 16];
        o[t] = wmma_bf16(p0, v0, o[t]);
        o[t] = wmma_bf16(p1, v1, o[t]);
      }
    }
    __syncthreads();
  }

  const int rb = (lane >> 4) * 8, cn = lane & 15;
#pragma unroll
  for (int r = 0; r < 8; ++r) {
    float inv = 1.f / l_run[r];
    int gq = q0 + w * 16 + rb + r;
    size_t base = ((size_t)b * N_QUERIES + gq) * D_MODEL + h * DIM_HEAD;
#pragma unroll
    for (int t = 0; t < 4; ++t)
      attn_out[base + t * 16 + cn] = f2bf(o[t][r] * inv);
  }
}

// ---------------------------------------------------------------------------
extern "C" void kernel_launch(void* const* d_in, const int* in_sizes, int n_in,
                              void* d_out, int out_size, void* d_ws,
                              size_t ws_size, hipStream_t stream) {
  const float* x      = (const float*)d_in[0];
  const float* query  = (const float*)d_in[1];
  const float* ln_q_w = (const float*)d_in[2];
  const float* ln_q_b = (const float*)d_in[3];
  const float* ln_k_w = (const float*)d_in[4];
  const float* ln_k_b = (const float*)d_in[5];
  const float* Wq     = (const float*)d_in[6];
  const float* Wkv    = (const float*)d_in[7];
  const float* Wout   = (const float*)d_in[8];
  float* out = (float*)d_out;

  const int ROWS = BATCH * SEQ;                 // 32768
  char* ws = (char*)d_ws;
  size_t off = 0;
  auto alloc = [&](size_t bytes) -> void* {
    void* p = ws + off;
    off += (bytes + 255) & ~(size_t)255;
    return p;
  };
  float*          stats  = (float*)alloc((size_t)ROWS * 2 * sizeof(float));
  unsigned short* qln    = (unsigned short*)alloc((size_t)N_QUERIES * D_MODEL * 2);
  unsigned short* WqT    = (unsigned short*)alloc((size_t)D_MODEL * D_MODEL * 2);
  unsigned short* WkvT   = (unsigned short*)alloc((size_t)CTX_DIM * KV_W * 2);
  unsigned short* WoutT  = (unsigned short*)alloc((size_t)D_MODEL * D_MODEL * 2);
  unsigned short* qproj  = (unsigned short*)alloc((size_t)N_QUERIES * D_MODEL * 2);
  unsigned short* kbuf   = (unsigned short*)alloc((size_t)ROWS * D_MODEL * 2);
  unsigned short* vtbuf  = (unsigned short*)alloc((size_t)ROWS * D_MODEL * 2);
  unsigned short* attnb  = (unsigned short*)alloc((size_t)BATCH * N_QUERIES * D_MODEL * 2);

  // 1) LN stats for x rows
  rowstats_kernel<<<ROWS, 256, 0, stream>>>(x, stats);
  // 2) LN(query) -> bf16
  qln_kernel<<<N_QUERIES, 256, 0, stream>>>(query, ln_q_w, ln_q_b, qln);
  // 3) weight converts (transposed)
  {
    int n1 = D_MODEL * D_MODEL;
    int n2 = CTX_DIM * KV_W;
    cvt_t_kernel<<<(n1 + 255) / 256, 256, 0, stream>>>(Wq,   WqT,   D_MODEL, D_MODEL);
    cvt_t_kernel<<<(n2 + 255) / 256, 256, 0, stream>>>(Wkv,  WkvT,  CTX_DIM, KV_W);
    cvt_t_kernel<<<(n1 + 255) / 256, 256, 0, stream>>>(Wout, WoutT, D_MODEL, D_MODEL);
  }
  // 4) q_proj = qln @ Wq  (256 x 768 x 768), bf16 out
  gemm_bf16_kernel<false><<<dim3(N_QUERIES / 128, D_MODEL / 128), 256, 0,
                            stream>>>(qln, WqT, (void*)qproj,
                                      N_QUERIES, D_MODEL, D_MODEL);
  // 5) kv = LN(x) @ Wkv  (32768 x 1536 x 1024); K row-major, V transposed
  gemm_lnx_kernel<<<dim3(ROWS / 128, KV_W / 128), 256, 0, stream>>>(
      x, stats, ln_k_w, ln_k_b, WkvT, kbuf, vtbuf);
  // 6) attention (flash, online softmax)
  attn_kernel<<<dim3(BATCH * N_HEAD, N_QUERIES / 128), 256, 0, stream>>>(
      qproj, kbuf, vtbuf, attnb);
  // 7) out = attn @ Wout  (2048 x 768 x 768), fp32 out
  gemm_bf16_kernel<true><<<dim3(BATCH * N_QUERIES / 128, D_MODEL / 128), 256, 0,
                           stream>>>(attnb, WoutT, (void*)out,
                                     BATCH * N_QUERIES, D_MODEL, D_MODEL);
}
